// GCNRegressor_6012954214846
// MI455X (gfx1250) — matrix operand order repaired and checked by reading, not compiled, not run on hardware
//
#include <hip/hip_runtime.h>
#include <math.h>

#define Bt 16
#define Nn 2048
#define Fd 64
#define KPAD 8
#define LROW (Nn + KPAD)   // 2056 f16 per LDS row (padding spreads banks)

typedef __attribute__((ext_vector_type(16))) _Float16 v16h;
typedef __attribute__((ext_vector_type(8)))  _Float16 v8h;
typedef __attribute__((ext_vector_type(2)))  __fp16   h2;   // matches cvt_pkrtz return
typedef __attribute__((ext_vector_type(8)))  float    v8f;

// ---------------------------------------------------------------------------
// Kernel 1: degree = rowsum(adj) + 1e-8.  One wave (32 lanes) per row.
// ---------------------------------------------------------------------------
__global__ void deg_kernel(const float* __restrict__ adj, float* __restrict__ deg) {
  const int wv = threadIdx.x >> 5, lane = threadIdx.x & 31;
  const int row = blockIdx.x * 8 + wv;                 // row in [0, B*N)
  const float* ap = adj + (size_t)row * Nn;
  float s = 0.f;
  for (int c = lane * 4; c < Nn; c += 128) {
    float4 v = *(const float4*)(ap + c);
    s += v.x + v.y + v.z + v.w;
  }
#pragma unroll
  for (int o = 16; o > 0; o >>= 1) s += __shfl_down(s, o, 32);
  if (lane == 0) deg[row] = s + 1e-8f;
}

// ---------------------------------------------------------------------------
// Kernel 2: projection h = x @ W + b, written TRANSPOSED as f16: hT[b][d][n].
// One thread per node; W cached in LDS (broadcast reads).
// ---------------------------------------------------------------------------
__global__ void proj_kernel(const float* __restrict__ xin,
                            const float* __restrict__ W,
                            const float* __restrict__ bias,
                            _Float16* __restrict__ hT) {
  __shared__ float sW[Fd * Fd];
  __shared__ float sBias[Fd];
  for (int i = threadIdx.x; i < Fd * Fd; i += blockDim.x) sW[i] = W[i];
  if (threadIdx.x < Fd) sBias[threadIdx.x] = bias[threadIdx.x];
  __syncthreads();

  const int g = blockIdx.x * blockDim.x + threadIdx.x;  // global node id
  const int b = g >> 11, n = g & (Nn - 1);
  float xr[Fd];
  const float* xp = xin + (size_t)g * Fd;
#pragma unroll
  for (int i = 0; i < Fd / 4; ++i) {
    float4 v = *(const float4*)(xp + 4 * i);
    xr[4 * i] = v.x; xr[4 * i + 1] = v.y; xr[4 * i + 2] = v.z; xr[4 * i + 3] = v.w;
  }
  _Float16* hp = hT + (size_t)b * Fd * Nn + n;
  for (int d = 0; d < Fd; ++d) {
    float acc = sBias[d];
#pragma unroll 8
    for (int f = 0; f < Fd; ++f) acc = fmaf(xr[f], sW[f * Fd + d], acc);
    hp[(size_t)d * Nn] = (_Float16)acc;                 // coalesced along n
  }
}

// ---------------------------------------------------------------------------
// Kernel 3: aggregation  xout = relu((adj @ h) / deg) + xin
// Workgroup: 8 waves, 128 rows of one batch. Entire hT[b] (256 KB f16) staged
// in LDS (CDNA5: 320 KB/WGP). Each wave: 16 rows, K-loop over 2048 in steps
// of 32 with v_wmma_f32_16x16x32_f16 against 4 LDS-resident B tiles (N=64).
// ---------------------------------------------------------------------------
__global__ __launch_bounds__(256)
void agg_kernel(const float* __restrict__ adj,
                const _Float16* __restrict__ hTg,
                const float* __restrict__ deg,
                const float* __restrict__ xin,
                float* __restrict__ xout) {
  extern __shared__ _Float16 sB[];                      // [64][LROW]
  const int b  = blockIdx.x >> 4;
  const int mb = blockIdx.x & 15;

  // Stage hT[b] (64 x 2048 f16 = 256 KB) into LDS, 16B chunks.
  {
    const _Float16* src = hTg + (size_t)b * Fd * Nn;
    const int t = threadIdx.x;
#pragma unroll 4
    for (int j = 0; j < 64; ++j) {
      int c  = t + 256 * j;                             // chunk id, 0..16383
      int d  = c >> 8;                                  // feature row
      int kc = c & 255;                                 // 8-element chunk in row
      *(uint4*)(&sB[d * LROW + kc * 8]) =
          *(const uint4*)(src + (size_t)d * Nn + kc * 8);
    }
  }
  __syncthreads();

  const int wv = threadIdx.x >> 5;
  const int lane = threadIdx.x & 31;
  const int lhalf = lane >> 4;
  const int l15 = lane & 15;
  const int m0w = mb * 128 + wv * 16;
  const int row = m0w + l15;
  // A-fragment K offsets per ISA layout: lanes<16 -> K {0..7,16..23},
  // lanes>=16 -> K {8..15,24..31}
  const float* ap = adj + (size_t)(b * Nn + row) * Nn + (lhalf ? 8 : 0);
  const int kb0 = lhalf ? 8 : 0;

  const v8f z = {0.f, 0.f, 0.f, 0.f, 0.f, 0.f, 0.f, 0.f};
  v8f acc[4] = {z, z, z, z};

#pragma unroll 2
  for (int kk = 0; kk < Nn; kk += 32) {
    __builtin_prefetch(ap + kk + 256, 0, 0);            // stream adj ahead
    float4 fa = *(const float4*)(ap + kk);
    float4 fb = *(const float4*)(ap + kk + 4);
    float4 fc = *(const float4*)(ap + kk + 16);
    float4 fd = *(const float4*)(ap + kk + 20);
    union { v16h v; h2 p[8]; } ua;
    ua.p[0] = __builtin_amdgcn_cvt_pkrtz(fa.x, fa.y);
    ua.p[1] = __builtin_amdgcn_cvt_pkrtz(fa.z, fa.w);
    ua.p[2] = __builtin_amdgcn_cvt_pkrtz(fb.x, fb.y);
    ua.p[3] = __builtin_amdgcn_cvt_pkrtz(fb.z, fb.w);
    ua.p[4] = __builtin_amdgcn_cvt_pkrtz(fc.x, fc.y);
    ua.p[5] = __builtin_amdgcn_cvt_pkrtz(fc.z, fc.w);
    ua.p[6] = __builtin_amdgcn_cvt_pkrtz(fd.x, fd.y);
    ua.p[7] = __builtin_amdgcn_cvt_pkrtz(fd.z, fd.w);
    const int kb = kk + kb0;
#pragma unroll
    for (int nt = 0; nt < 4; ++nt) {
      union { v16h v; v8h h8[2]; } ub;                  // B: col n = nt*16+l15,
      const _Float16* bp = &sB[(nt * 16 + l15) * LROW + kb];  // K {8h..8h+7, 16+8h..16+8h+7}
      ub.h8[0] = *(const v8h*)(bp);
      ub.h8[1] = *(const v8h*)(bp + 16);
      acc[nt] = __builtin_amdgcn_wmma_f32_16x16x32_f16(
          false, ua.v, false, ub.v, (short)0, acc[nt], false, false);
    }
  }

  // Epilogue: C/D layout -> lane L: m = m0w + r + 8*lhalf, n = nt*16 + l15
  float dr[8];
  const int mrb = m0w + lhalf * 8;
#pragma unroll
  for (int r = 0; r < 8; ++r) dr[r] = deg[b * Nn + mrb + r];
  const size_t xbase = (size_t)b * Nn * Fd;
#pragma unroll
  for (int nt = 0; nt < 4; ++nt) {
    const int n = nt * 16 + l15;
#pragma unroll
    for (int r = 0; r < 8; ++r) {
      const int m = mrb + r;
      float v = fmaxf(acc[nt][r] / dr[r], 0.f);
      size_t idx = xbase + (size_t)m * Fd + n;
      xout[idx] = v + xin[idx];                         // residual
    }
  }
}

// ---------------------------------------------------------------------------
// Kernel 4: pooling + MLP head + softmax, one block per batch element.
// ---------------------------------------------------------------------------
__global__ void head_kernel(const float* __restrict__ x,
                            const float* __restrict__ ginfo,
                            const float* __restrict__ w1, const float* __restrict__ b1,
                            const float* __restrict__ w2, const float* __restrict__ b2,
                            const float* __restrict__ w3, const float* __restrict__ b3,
                            float* __restrict__ out) {
  __shared__ float part[4][64];
  __shared__ float zs[80];
  __shared__ float y1[256];
  __shared__ float y2[128];
  __shared__ float lg[200];
  __shared__ float red[2];
  const int b = blockIdx.x;
  const int t = threadIdx.x;
  const int f = t & 63, p = t >> 6;
  float s = 0.f;
  for (int n = p; n < Nn; n += 4) s += x[((size_t)b * Nn + n) * Fd + f];
  part[p][f] = s;
  __syncthreads();
  if (t < 64)      zs[t] = part[0][t] + part[1][t] + part[2][t] + part[3][t];
  else if (t < 80) zs[t] = ginfo[b * 16 + (t - 64)];
  __syncthreads();
  {
    float a = b1[t];
    for (int j = 0; j < 80; ++j) a = fmaf(zs[j], w1[j * 256 + t], a);
    y1[t] = fmaxf(a, 0.f);
  }
  __syncthreads();
  if (t < 128) {
    float a = b2[t];
    for (int j = 0; j < 256; ++j) a = fmaf(y1[j], w2[j * 128 + t], a);
    y2[t] = fmaxf(a, 0.f);
  }
  __syncthreads();
  if (t < 200) {
    float a = b3[t];
    for (int j = 0; j < 128; ++j) a = fmaf(y2[j], w3[j * 200 + t], a);
    lg[t] = a;
  }
  __syncthreads();
  if (t == 0) {
    float mx = lg[0];
    for (int j = 1; j < 200; ++j) mx = fmaxf(mx, lg[j]);
    float sum = 0.f;
    for (int j = 0; j < 200; ++j) sum += __expf(lg[j] - mx);
    red[0] = mx; red[1] = sum;
  }
  __syncthreads();
  if (t < 200) out[b * 200 + t] = __expf(lg[t] - red[0]) / red[1];
}

// ---------------------------------------------------------------------------
extern "C" void kernel_launch(void* const* d_in, const int* in_sizes, int n_in,
                              void* d_out, int out_size, void* d_ws, size_t ws_size,
                              hipStream_t stream) {
  const float* node  = (const float*)d_in[0];
  const float* adj   = (const float*)d_in[1];
  const float* ginfo = (const float*)d_in[2];
  const float* gw    = (const float*)d_in[3];
  const float* gb    = (const float*)d_in[4];
  const float* w1    = (const float*)d_in[5];
  const float* b1    = (const float*)d_in[6];
  const float* w2    = (const float*)d_in[7];
  const float* b2    = (const float*)d_in[8];
  const float* w3    = (const float*)d_in[9];
  const float* b3    = (const float*)d_in[10];
  float* out = (float*)d_out;

  char* ws = (char*)d_ws;
  float*    deg = (float*)(ws);                                // 128 KB
  float*    x0  = (float*)(ws + 131072);                       // 8 MB
  float*    x1  = (float*)(ws + 131072 + 8388608);             // 8 MB
  _Float16* hT  = (_Float16*)(ws + 131072 + 2 * 8388608);      // 4 MB

  deg_kernel<<<Bt * Nn / 8, 256, 0, stream>>>(adj, deg);

  const float* xin_l[3]  = { node, x0, x1 };
  float*       xout_l[3] = { x0,   x1, x0 };
  for (int L = 0; L < 3; ++L) {
    proj_kernel<<<Bt * Nn / 128, 128, 0, stream>>>(
        xin_l[L], gw + L * Fd * Fd, gb + L * Fd, hT);
    agg_kernel<<<Bt * 16, 256, (size_t)Fd * LROW * sizeof(_Float16), stream>>>(
        adj, hT, deg, xin_l[L], xout_l[L]);
  }
  head_kernel<<<Bt, 256, 0, stream>>>(x0, ginfo, w1, b1, w2, b2, w3, b3, out);
}
